// DualSelfAttention_11639361372450
// MI455X (gfx1250) — matrix operand order repaired and checked by reading, not compiled, not run on hardware
//
#include <hip/hip_runtime.h>
#include <cstdint>
#include <cstddef>

// Problem constants (match reference)
#define BB 4
#define CC 512
#define C8 64
#define NN 4096

typedef __attribute__((ext_vector_type(16))) __bf16 v16bf;
typedef __attribute__((ext_vector_type(8)))  float  v8f;
typedef __attribute__((ext_vector_type(4)))  unsigned int u32x4;

union V16U { v16bf v; u32x4 u[2]; };
union V8BF { u32x4 u; __bf16 e[8]; };

static __device__ __forceinline__ v8f zero8() {
  v8f z = {0.f, 0.f, 0.f, 0.f, 0.f, 0.f, 0.f, 0.f};
  return z;
}

static __device__ __forceinline__ v8f wmma_bf(v16bf a, v16bf b, v8f c) {
  // D = A(16x32 bf16) * B(32x16 bf16) + C(16x16 f32)
  return __builtin_amdgcn_wmma_f32_16x16x32_bf16(
      /*neg_a=*/false, a, /*neg_b=*/false, b,
      /*c_mod=*/(short)0, c, /*reuse_a=*/false, /*reuse_b=*/false);
}

// A operand tile (16M x 32K bf16). Storage: element(m,k) at base[m*ldk + k].
// Layout (ISA 7.12.2): lanes 0-15 row m, K=[k0..k0+7] in v0-3, K=[k0+16..k0+23] in v4-7;
// lanes 16-31 same rows, K offset +8.
static __device__ __forceinline__ v16bf load_a16(const __bf16* base, int ldk,
                                                 int m0, int k0, int lane) {
  const __bf16* p = base + (size_t)(m0 + (lane & 15)) * ldk + k0 + ((lane >> 4) << 3);
  V16U r;
  r.u[0] = *(const u32x4*)(p);
  r.u[1] = *(const u32x4*)(p + 16);
  return r.v;
}

// B operand tile (32K x 16N bf16). Storage: element(k,n) at base[n*ldk + k] (B^T, K-major).
// Layout: lanes 0-15 col n, K=[k0..k0+7] in v0-3, K=[k0+16..k0+23] in v4-7; lanes 16-31 same cols, K offset +8.
static __device__ __forceinline__ v16bf load_b16(const __bf16* base, int ldk,
                                                 int n0, int k0, int lane) {
  const __bf16* p = base + (size_t)(n0 + (lane & 15)) * ldk + k0 + ((lane >> 4) << 3);
  V16U r;
  r.u[0] = *(const u32x4*)(p);
  r.u[1] = *(const u32x4*)(p + 16);
  return r.v;
}

// ---------------------------------------------------------------------------
// Elementwise f32 -> bf16 (weights)
__global__ __launch_bounds__(256) void cvt_bf16_kernel(const float* __restrict__ in,
                                                       __bf16* __restrict__ out, int n) {
  int i = blockIdx.x * 256 + threadIdx.x;
  if (i < n) out[i] = (__bf16)in[i];
}

// Tiled transpose + convert: in f32 (R,S) row-major -> out bf16 (S,R), per batch z.
__global__ __launch_bounds__(256) void transpose_cvt_kernel(const float* __restrict__ in,
                                                            __bf16* __restrict__ out,
                                                            int R, int S) {
  __shared__ float t[32][33];
  const int b = blockIdx.z;
  const float* inb = in + (size_t)b * R * S;
  __bf16* outb = out + (size_t)b * R * S;
  const int s0 = blockIdx.x * 32, r0 = blockIdx.y * 32;
  const int tx = threadIdx.x, ty = threadIdx.y;   // blockDim (32,8)
#pragma unroll
  for (int i = 0; i < 4; ++i)
    t[ty + 8 * i][tx] = inb[(size_t)(r0 + ty + 8 * i) * S + s0 + tx];
  __syncthreads();
#pragma unroll
  for (int i = 0; i < 4; ++i)
    outb[(size_t)(s0 + ty + 8 * i) * R + r0 + tx] = (__bf16)t[tx][ty + 8 * i];
}

// ---------------------------------------------------------------------------
// Projection GEMM: Y(Cout x NN) = W(Cout x 512) * X(512 x NN) + bias.
// W: bf16 row-major (Cout,512). X provided transposed: xT bf16 (NN,512).
// mode 0: write Y bf16 row-major (Cout,NN); mode 1: write Y^T bf16 (NN,Cout).
// Each wave: 16 out-rows x 64 cols. blockDim=128 (4 waves).
__global__ __launch_bounds__(128) void proj_gemm_kernel(const __bf16* __restrict__ W,
                                                        const float* __restrict__ bias,
                                                        const __bf16* __restrict__ xT,
                                                        __bf16* __restrict__ out,
                                                        int Cout, int mode) {
  const int lane = threadIdx.x & 31;
  const int wave = threadIdx.x >> 5;
  const int b = blockIdx.z;
  const int m0 = blockIdx.y * 16;
  const int n0 = blockIdx.x * 256 + wave * 64;
  const __bf16* xb = xT + (size_t)b * NN * CC;
  __bf16* ob = out + (size_t)b * (size_t)Cout * NN;

  v8f acc[4];
#pragma unroll
  for (int t = 0; t < 4; ++t) acc[t] = zero8();

  for (int k = 0; k < CC; k += 32) {
    if (k + 32 < CC)
      __builtin_prefetch((const void*)(xb + (size_t)(n0 + (lane & 15)) * CC + k + 32), 0, 1);
    v16bf a = load_a16(W, CC, m0, k, lane);
#pragma unroll
    for (int t = 0; t < 4; ++t) {
      v16bf bt = load_b16(xb, CC, n0 + t * 16, k, lane);
      acc[t] = wmma_bf(a, bt, acc[t]);
    }
  }

  const int hi8 = (lane >> 4) << 3;
  float bv[8];
#pragma unroll
  for (int v = 0; v < 8; ++v) bv[v] = bias[m0 + v + hi8];

  if (mode == 0) {
    // row-major (Cout, NN): coalesced across lanes per vgpr row
#pragma unroll
    for (int t = 0; t < 4; ++t) {
      int n = n0 + t * 16 + (lane & 15);
#pragma unroll
      for (int v = 0; v < 8; ++v) {
        int m = m0 + v + hi8;
        ob[(size_t)m * NN + n] = (__bf16)(acc[t][v] + bv[v]);
      }
    }
  } else {
    // transposed (NN, Cout): contiguous 8 bf16 per lane -> 16B stores
#pragma unroll
    for (int t = 0; t < 4; ++t) {
      int n = n0 + t * 16 + (lane & 15);
      V8BF r;
#pragma unroll
      for (int v = 0; v < 8; ++v) r.e[v] = (__bf16)(acc[t][v] + bv[v]);
      *(u32x4*)&ob[(size_t)n * Cout + m0 + hi8] = r.u;
    }
  }
}

// ---------------------------------------------------------------------------
// Position-attention softmax statistics: S = Q^T K (q x k), row max m and sum l.
// qT,kT: bf16 (NN, 64). Outputs mOut, linvOut: f32 (BB, NN).
// blockDim=256 (8 waves), one 16-query tile per wave.
__global__ __launch_bounds__(256) void pos_stats_kernel(const __bf16* __restrict__ qT,
                                                        const __bf16* __restrict__ kT,
                                                        float* __restrict__ mOut,
                                                        float* __restrict__ linvOut) {
  const int lane = threadIdx.x & 31;
  const int wave = threadIdx.x >> 5;
  const int b = blockIdx.y;
  const int q0 = (blockIdx.x * 8 + wave) * 16;
  const __bf16* qb = qT + (size_t)b * NN * C8;
  const __bf16* kb = kT + (size_t)b * NN * C8;

  v16bf a0 = load_a16(qb, C8, q0, 0, lane);
  v16bf a1 = load_a16(qb, C8, q0, 32, lane);

  float m[8], l[8];
#pragma unroll
  for (int v = 0; v < 8; ++v) { m[v] = -1e30f; l[v] = 0.f; }

  for (int kt = 0; kt < NN / 16; ++kt) {
    v16bf b0 = load_b16(kb, C8, kt * 16, 0, lane);
    v16bf b1 = load_b16(kb, C8, kt * 16, 32, lane);
    v8f s = zero8();
    s = wmma_bf(a0, b0, s);
    s = wmma_bf(a1, b1, s);
#pragma unroll
    for (int v = 0; v < 8; ++v) {
      float sv = s[v];
      float mn = fmaxf(m[v], sv);
      l[v] = l[v] * __expf(m[v] - mn) + __expf(sv - mn);
      m[v] = mn;
    }
  }
  // merge across the 16 lanes of each half (rows 0-7 live in lanes 0-15, 8-15 in 16-31)
#pragma unroll
  for (int v = 0; v < 8; ++v) {
#pragma unroll
    for (int off = 1; off < 16; off <<= 1) {
      float om = __shfl_xor(m[v], off, 32);
      float ol = __shfl_xor(l[v], off, 32);
      float mn = fmaxf(m[v], om);
      l[v] = l[v] * __expf(m[v] - mn) + ol * __expf(om - mn);
      m[v] = mn;
    }
  }
  const int hi8 = (lane >> 4) << 3;
  if ((lane & 15) == 0) {
#pragma unroll
    for (int v = 0; v < 8; ++v) {
      int r = q0 + v + hi8;
      mOut[b * NN + r] = m[v];
      linvOut[b * NN + r] = 1.0f / l[v];
    }
  }
}

// ---------------------------------------------------------------------------
// Position attention PV with fused residual:
// posOut(C,N) = gpos * ( V(C,N) x P^T(N,Nq) ) + x(C,N)
// P recomputed tile-by-tile into LDS (16 queries x 1024-k chunk, bf16, A layout).
// blockDim=128 (4 waves): phase1 all waves fill sP, phase2 each wave owns 128 channels.
#define PV_KCHUNK 1024
#define PV_PAD 8
__global__ __launch_bounds__(128) void pos_pv_kernel(const __bf16* __restrict__ qT,
                                                     const __bf16* __restrict__ kT,
                                                     const __bf16* __restrict__ V,
                                                     const float* __restrict__ mArr,
                                                     const float* __restrict__ linvArr,
                                                     const float* __restrict__ x,
                                                     const float* __restrict__ gpos,
                                                     float* __restrict__ posOut) {
  __shared__ __bf16 sP[16][PV_KCHUNK + PV_PAD];
  const int lane = threadIdx.x & 31;
  const int wave = threadIdx.x >> 5;
  const int b = blockIdx.y;
  const int q0 = blockIdx.x * 16;
  const int hi8 = (lane >> 4) << 3;

  const __bf16* qb = qT + (size_t)b * NN * C8;
  const __bf16* kb = kT + (size_t)b * NN * C8;
  const __bf16* vb = V + (size_t)b * CC * NN;

  float mrow[8], lrow[8];
#pragma unroll
  for (int v = 0; v < 8; ++v) {
    int r = q0 + v + hi8;
    mrow[v] = mArr[b * NN + r];
    lrow[v] = linvArr[b * NN + r];
  }
  v16bf aq0 = load_a16(qb, C8, q0, 0, lane);
  v16bf aq1 = load_a16(qb, C8, q0, 32, lane);

  v8f acc[8];
#pragma unroll
  for (int t = 0; t < 8; ++t) acc[t] = zero8();

  const int c0w = wave * 128;

  for (int chunk = 0; chunk < NN / PV_KCHUNK; ++chunk) {
    // ---- phase 1: scores -> P (bf16) in LDS ----
    for (int t = 0; t < (PV_KCHUNK / 16) / 4; ++t) {   // 16 score tiles per wave
      int kg = (chunk * (PV_KCHUNK / 16) + wave * ((PV_KCHUNK / 16) / 4) + t) * 16;
      v16bf b0 = load_b16(kb, C8, kg, 0, lane);
      v16bf b1 = load_b16(kb, C8, kg, 32, lane);
      v8f s = zero8();
      s = wmma_bf(aq0, b0, s);
      s = wmma_bf(aq1, b1, s);
      int klocal = (kg - chunk * PV_KCHUNK) + (lane & 15);
#pragma unroll
      for (int v = 0; v < 8; ++v) {
        float p = __expf(s[v] - mrow[v]) * lrow[v];
        sP[v + hi8][klocal] = (__bf16)p;
      }
    }
    __syncthreads();
    // ---- phase 2: acc += V * P^T for this k chunk ----
    for (int ks = 0; ks < PV_KCHUNK / 32; ++ks) {
      const int kl = ks * 32;
      const __bf16* pA = &sP[lane & 15][kl + hi8];
      V16U a;
      a.u[0] = *(const u32x4*)(pA);
      a.u[1] = *(const u32x4*)(pA + 16);
      const int kgl = chunk * PV_KCHUNK + kl;
#pragma unroll
      for (int t = 0; t < 8; ++t) {
        v16bf bv = load_b16(vb, NN, c0w + t * 16, kgl, lane);
        acc[t] = wmma_bf(a.v, bv, acc[t]);
      }
    }
    __syncthreads();
  }

  // ---- epilogue: posOut = gpos*acc + x, store (C,N) f32 ----
  const float g = gpos[0];
#pragma unroll
  for (int t = 0; t < 8; ++t) {
    int c = c0w + t * 16 + (lane & 15);
    size_t base = ((size_t)(b * CC + c)) * NN + q0 + hi8;
    float4 x0 = *(const float4*)&x[base];
    float4 x1 = *(const float4*)&x[base + 4];
    float4 o0, o1;
    o0.x = g * acc[t][0] + x0.x;  o0.y = g * acc[t][1] + x0.y;
    o0.z = g * acc[t][2] + x0.z;  o0.w = g * acc[t][3] + x0.w;
    o1.x = g * acc[t][4] + x1.x;  o1.y = g * acc[t][5] + x1.y;
    o1.z = g * acc[t][6] + x1.z;  o1.w = g * acc[t][7] + x1.w;
    *(float4*)&posOut[base] = o0;
    *(float4*)&posOut[base + 4] = o1;
  }
}

// ---------------------------------------------------------------------------
// Channel scores: S(i,j) = sum_n Q(i,n) K(j,n).  Q,K bf16 (C,N) row-major.
// One wave per 32x32 output tile.
__global__ __launch_bounds__(32) void chan_s_kernel(const __bf16* __restrict__ Q,
                                                    const __bf16* __restrict__ K,
                                                    float* __restrict__ S) {
  const int lane = threadIdx.x & 31;
  const int b = blockIdx.z;
  const int i0 = blockIdx.y * 32, j0 = blockIdx.x * 32;
  const __bf16* qb = Q + (size_t)b * CC * NN;
  const __bf16* kb = K + (size_t)b * CC * NN;

  v8f acc[2][2];
#pragma unroll
  for (int a = 0; a < 2; ++a)
#pragma unroll
    for (int c = 0; c < 2; ++c) acc[a][c] = zero8();

  for (int k = 0; k < NN; k += 32) {
    if (k + 32 < NN)
      __builtin_prefetch((const void*)(qb + (size_t)(i0 + (lane & 15)) * NN + k + 32), 0, 1);
    v16bf a0 = load_a16(qb, NN, i0, k, lane);
    v16bf a1 = load_a16(qb, NN, i0 + 16, k, lane);
    v16bf b0 = load_b16(kb, NN, j0, k, lane);
    v16bf b1 = load_b16(kb, NN, j0 + 16, k, lane);
    acc[0][0] = wmma_bf(a0, b0, acc[0][0]);
    acc[0][1] = wmma_bf(a0, b1, acc[0][1]);
    acc[1][0] = wmma_bf(a1, b0, acc[1][0]);
    acc[1][1] = wmma_bf(a1, b1, acc[1][1]);
  }
  const int hi8 = (lane >> 4) << 3;
#pragma unroll
  for (int ti = 0; ti < 2; ++ti)
#pragma unroll
    for (int tj = 0; tj < 2; ++tj) {
      int j = j0 + tj * 16 + (lane & 15);
#pragma unroll
      for (int v = 0; v < 8; ++v) {
        int i = i0 + ti * 16 + v + hi8;
        S[((size_t)(b * CC + i)) * CC + j] = acc[ti][tj][v];
      }
    }
}

// Row softmax over 512 values; write bf16 P (C,C) row-major. One wave per row.
__global__ __launch_bounds__(256) void chan_softmax_kernel(const float* __restrict__ S,
                                                           __bf16* __restrict__ P) {
  const int lane = threadIdx.x & 31;
  const int wave = threadIdx.x >> 5;
  const int b = blockIdx.y;
  const int row = blockIdx.x * 8 + wave;
  const float* src = S + ((size_t)(b * CC + row)) * CC;
  __bf16* dst = P + ((size_t)(b * CC + row)) * CC;

  float v[16];
  float mx = -1e30f;
#pragma unroll
  for (int t = 0; t < 16; ++t) { v[t] = src[lane + 32 * t]; mx = fmaxf(mx, v[t]); }
#pragma unroll
  for (int off = 16; off > 0; off >>= 1) mx = fmaxf(mx, __shfl_xor(mx, off, 32));
  float sum = 0.f;
#pragma unroll
  for (int t = 0; t < 16; ++t) { v[t] = __expf(v[t] - mx); sum += v[t]; }
#pragma unroll
  for (int off = 16; off > 0; off >>= 1) sum += __shfl_xor(sum, off, 32);
  const float inv = 1.0f / sum;
#pragma unroll
  for (int t = 0; t < 16; ++t) dst[lane + 32 * t] = (__bf16)(v[t] * inv);
}

// Final: out(i,n) = gchan * sum_j P(i,j) Vc(j,n) + posOut(i,n).
// P bf16 (C,C) row-major; Vc provided transposed: vT bf16 (N,C).
// blockDim=128 (4 waves), each wave 16 rows x 64 cols.
__global__ __launch_bounds__(128) void chan_out_kernel(const __bf16* __restrict__ P,
                                                       const __bf16* __restrict__ vT,
                                                       const float* __restrict__ posOut,
                                                       const float* __restrict__ gchan,
                                                       float* __restrict__ out) {
  const int lane = threadIdx.x & 31;
  const int wave = threadIdx.x >> 5;
  const int b = blockIdx.z;
  const int i0 = (blockIdx.y * 4 + wave) * 16;
  const int n0 = blockIdx.x * 64;
  const __bf16* pb = P + (size_t)b * CC * CC;
  const __bf16* vb = vT + (size_t)b * NN * CC;

  v8f acc[4];
#pragma unroll
  for (int t = 0; t < 4; ++t) acc[t] = zero8();

  for (int k = 0; k < CC; k += 32) {
    v16bf a = load_a16(pb, CC, i0, k, lane);
#pragma unroll
    for (int t = 0; t < 4; ++t) {
      v16bf bt = load_b16(vb, CC, n0 + t * 16, k, lane);
      acc[t] = wmma_bf(a, bt, acc[t]);
    }
  }
  const int hi8 = (lane >> 4) << 3;
  const float g = gchan[0];
#pragma unroll
  for (int t = 0; t < 4; ++t) {
    int n = n0 + t * 16 + (lane & 15);
#pragma unroll
    for (int v = 0; v < 8; ++v) {
      int i = i0 + v + hi8;
      size_t idx = ((size_t)(b * CC + i)) * NN + n;
      out[idx] = g * acc[t][v] + posOut[idx];
    }
  }
}

// ---------------------------------------------------------------------------
extern "C" void kernel_launch(void* const* d_in, const int* in_sizes, int n_in,
                              void* d_out, int out_size, void* d_ws, size_t ws_size,
                              hipStream_t stream) {
  (void)in_sizes; (void)n_in; (void)out_size; (void)ws_size;
  const float* x     = (const float*)d_in[0];
  const float* wpq   = (const float*)d_in[1];
  const float* bpq   = (const float*)d_in[2];
  const float* wpk   = (const float*)d_in[3];
  const float* bpk   = (const float*)d_in[4];
  const float* wpv   = (const float*)d_in[5];
  const float* bpv   = (const float*)d_in[6];
  const float* wcq   = (const float*)d_in[7];
  const float* bcq   = (const float*)d_in[8];
  const float* wck   = (const float*)d_in[9];
  const float* bck   = (const float*)d_in[10];
  const float* wcv   = (const float*)d_in[11];
  const float* bcv   = (const float*)d_in[12];
  const float* gpos  = (const float*)d_in[13];
  const float* gchan = (const float*)d_in[14];
  float* out = (float*)d_out;

  char* ws = (char*)d_ws;
  size_t off = 0;
  auto take = [&](size_t bytes) -> char* {
    char* p = ws + off;
    off += (bytes + 255) & ~(size_t)255;
    return p;
  };

  __bf16* xT       = (__bf16*)take((size_t)BB * NN * CC * 2);   // x^T bf16 (B,N,C)
  __bf16* wpq_b    = (__bf16*)take((size_t)C8 * CC * 2);
  __bf16* wpk_b    = (__bf16*)take((size_t)C8 * CC * 2);
  __bf16* wpv_b    = (__bf16*)take((size_t)CC * CC * 2);
  __bf16* wcq_b    = (__bf16*)take((size_t)CC * CC * 2);
  __bf16* wck_b    = (__bf16*)take((size_t)CC * CC * 2);
  __bf16* wcv_b    = (__bf16*)take((size_t)CC * CC * 2);
  __bf16* pos_qT   = (__bf16*)take((size_t)BB * NN * C8 * 2);   // (B,N,64)
  __bf16* pos_kT   = (__bf16*)take((size_t)BB * NN * C8 * 2);   // (B,N,64)
  __bf16* pos_v    = (__bf16*)take((size_t)BB * CC * NN * 2);   // (B,C,N)
  float*  mArr     = (float*) take((size_t)BB * NN * 4);
  float*  linvArr  = (float*) take((size_t)BB * NN * 4);
  float*  pos_out  = (float*) take((size_t)BB * CC * NN * 4);   // (B,C,N)
  __bf16* pos_outT = (__bf16*)take((size_t)BB * NN * CC * 2);   // (B,N,C)
  __bf16* chan_q   = (__bf16*)take((size_t)BB * CC * NN * 2);   // (B,C,N)
  __bf16* chan_k   = (__bf16*)take((size_t)BB * CC * NN * 2);   // (B,C,N)
  __bf16* chan_vT  = (__bf16*)take((size_t)BB * NN * CC * 2);   // (B,N,C)
  float*  chan_S   = (float*) take((size_t)BB * CC * CC * 4);
  __bf16* chan_P   = (__bf16*)take((size_t)BB * CC * CC * 2);

  // 1) weights -> bf16
  cvt_bf16_kernel<<<dim3((C8 * CC) / 256), 256, 0, stream>>>(wpq, wpq_b, C8 * CC);
  cvt_bf16_kernel<<<dim3((C8 * CC) / 256), 256, 0, stream>>>(wpk, wpk_b, C8 * CC);
  cvt_bf16_kernel<<<dim3((CC * CC) / 256), 256, 0, stream>>>(wpv, wpv_b, CC * CC);
  cvt_bf16_kernel<<<dim3((CC * CC) / 256), 256, 0, stream>>>(wcq, wcq_b, CC * CC);
  cvt_bf16_kernel<<<dim3((CC * CC) / 256), 256, 0, stream>>>(wck, wck_b, CC * CC);
  cvt_bf16_kernel<<<dim3((CC * CC) / 256), 256, 0, stream>>>(wcv, wcv_b, CC * CC);

  // 2) x -> x^T bf16
  transpose_cvt_kernel<<<dim3(NN / 32, CC / 32, BB), dim3(32, 8), 0, stream>>>(x, xT, CC, NN);

  // 3) position projections
  proj_gemm_kernel<<<dim3(NN / 256, C8 / 16, BB), 128, 0, stream>>>(wpq_b, bpq, xT, pos_qT, C8, 1);
  proj_gemm_kernel<<<dim3(NN / 256, C8 / 16, BB), 128, 0, stream>>>(wpk_b, bpk, xT, pos_kT, C8, 1);
  proj_gemm_kernel<<<dim3(NN / 256, CC / 16, BB), 128, 0, stream>>>(wpv_b, bpv, xT, pos_v, CC, 0);

  // 4) softmax stats for position attention
  pos_stats_kernel<<<dim3(NN / 128, BB), 256, 0, stream>>>(pos_qT, pos_kT, mArr, linvArr);

  // 5) PV + residual -> pos_out
  pos_pv_kernel<<<dim3(NN / 16, BB), 128, 0, stream>>>(pos_qT, pos_kT, pos_v, mArr, linvArr,
                                                       x, gpos, pos_out);

  // 6) pos_out -> transposed bf16 for channel projections
  transpose_cvt_kernel<<<dim3(NN / 32, CC / 32, BB), dim3(32, 8), 0, stream>>>(pos_out, pos_outT,
                                                                               CC, NN);

  // 7) channel projections
  proj_gemm_kernel<<<dim3(NN / 256, CC / 16, BB), 128, 0, stream>>>(wcq_b, bcq, pos_outT, chan_q, CC, 0);
  proj_gemm_kernel<<<dim3(NN / 256, CC / 16, BB), 128, 0, stream>>>(wck_b, bck, pos_outT, chan_k, CC, 0);
  proj_gemm_kernel<<<dim3(NN / 256, CC / 16, BB), 128, 0, stream>>>(wcv_b, bcv, pos_outT, chan_vT, CC, 1);

  // 8) channel scores
  chan_s_kernel<<<dim3(CC / 32, CC / 32, BB), 32, 0, stream>>>(chan_q, chan_k, chan_S);

  // 9) channel softmax
  chan_softmax_kernel<<<dim3(CC / 8, BB), 256, 0, stream>>>(chan_S, chan_P);

  // 10) P @ V + residual -> output
  chan_out_kernel<<<dim3(NN / 64, CC / 64, BB), 128, 0, stream>>>(chan_P, chan_vT, pos_out,
                                                                  gchan, out);
}
